// SelfAttention_49220325212476
// MI455X (gfx1250) — matrix operand order repaired and checked by reading, not compiled, not run on hardware
//
#include <hip/hip_runtime.h>
#include <hip/hip_bf16.h>

// ---------------------------------------------------------------------------
// Fused pre-LN self-attention with score-LayerNorm, for MI455X (gfx1250).
// All matmuls via v_wmma_f32_16x16x32_f16 (f32 accumulate). Score matrix
// [B,H,S,S] (134 MB in f32) never leaves the WGP: it lives in LDS per
// (b,h,32-row) block. GEMMs are register-blocked 2x4 (m x n tiles) per wave
// with all per-K-step operands in distinct registers so loads pipeline.
// The V panel for the PV GEMM is staged to LDS with CDNA5 async loads
// (global_load_async_to_lds_b128, ASYNCcnt) overlapping QK^T + softmax.
// Problem sizes: B=16,S=512,D=512,H=8,DH=64.
// ---------------------------------------------------------------------------

typedef __attribute__((ext_vector_type(16))) _Float16 v16h;
typedef __attribute__((ext_vector_type(8)))  _Float16 v8h;
typedef __attribute__((ext_vector_type(8)))  float    v8f;

#define DEVFN static __device__ __forceinline__

constexpr int kB = 16, kS = 512, kD = 512, kH = 8, kDH = 64;
constexpr int kM = kB * kS;            // 8192 rows in [B*S, D] space
constexpr float kEPS = 1e-5f;
constexpr float kNEG = -1.0e9f;
constexpr float kQKScale = 0.125f;     // 1/sqrt(64)

// k_attn dynamic LDS layout (bytes)
constexpr unsigned kScoresOff   = 0;                     // [32][512] f32 = 64K
constexpr unsigned kProbsOff    = 65536;                 // [32][512] f16 = 32K
constexpr unsigned kRowScaleOff = 98304;                 // [32] f32
constexpr unsigned kVldsOff     = 98432;                 // [64][512] f16 = 64K
constexpr unsigned kSmemBytes   = 98432 + 65536;         // 163,968

// ---- WMMA fragment helpers (M/N per CDNA5 ISA 7.12.2; K order contiguous, same for A and B, wave32) ---------

// A fragment: 16x32 f16, row-major source with leading dim lda (halfs).
// lane l: M = l&15; elem e -> K = (l>=16)*16 + e.  One 32-byte load.
DEVFN v16h load_a_frag(const _Float16* A, int lda, int lane) {
  return *(const v16h*)(A + (lane & 15) * lda + ((lane >> 4) << 4));
}

// B fragment: 32x16 f16 taken from a TRANSPOSED operand Bt[N][K] (row-major,
// leading dim ldb in halfs), i.e. Bt[n][k] == B[k][n].
// lane l: N = l&15; elem e -> K = (l>=16)*16 + e.  One 32-byte load.
DEVFN v16h load_bt_frag(const _Float16* Bt, int ldb, int lane) {
  return *(const v16h*)(Bt + (lane & 15) * ldb + ((lane >> 4) << 4));
}

DEVFN v8f wmma_f16(v16h a, v16h b, v8f c) {
  return __builtin_amdgcn_wmma_f32_16x16x32_f16(
      /*neg_a=*/false, a, /*neg_b=*/false, b,
      /*c_mod=*/(short)0, c, /*reuse_a=*/false, /*reuse_b=*/false);
}

DEVFN float wave_sum(float v) {
  for (int o = 16; o > 0; o >>= 1) v += __shfl_xor(v, o, 32);
  return v;
}
DEVFN float wave_max(float v) {
  for (int o = 16; o > 0; o >>= 1) v = fmaxf(v, __shfl_xor(v, o, 32));
  return v;
}

// ---- Kernel 0: weights f32[K][N] -> f16 transposed Wt[N][K], 4 matrices ---
__global__ void k_prep_weights(const float* __restrict__ Wq,
                               const float* __restrict__ Wk,
                               const float* __restrict__ Wv,
                               const float* __restrict__ Wd,
                               _Float16* __restrict__ WtAll) {
  int idx = blockIdx.x * 256 + threadIdx.x;          // < 4 * 512 * 512
  int mat = idx >> 18;
  int e   = idx & 262143;
  int kk  = e >> 9;
  int nn  = e & 511;
  const float* W = (mat == 0) ? Wq : (mat == 1) ? Wk : (mat == 2) ? Wv : Wd;
  WtAll[mat * 262144 + nn * 512 + kk] = (_Float16)W[kk * 512 + nn];
}

// ---- Kernel 1: pre-LayerNorm over D, emit f16 h ---------------------------
__global__ void k_layernorm(const float* __restrict__ x,
                            const float* __restrict__ g,
                            const float* __restrict__ bta,
                            _Float16* __restrict__ h16) {
  int row = blockIdx.x;                // [0, B*S)
  int t = threadIdx.x;                 // 256 threads, D = 512
  int lane = t & 31, w = t >> 5;
  float v0 = x[row * kD + t];
  float v1 = x[row * kD + t + 256];
  float sum = v0 + v1;
  float sq  = v0 * v0 + v1 * v1;
  sum = wave_sum(sum);
  sq  = wave_sum(sq);
  __shared__ float s_sum[8], s_sq[8];
  if (lane == 0) { s_sum[w] = sum; s_sq[w] = sq; }
  __syncthreads();
  float S = 0.f, Q = 0.f;
  for (int i = 0; i < 8; ++i) { S += s_sum[i]; Q += s_sq[i]; }
  float mu  = S * (1.0f / kD);
  float var = Q * (1.0f / kD) - mu * mu;
  float rs  = rsqrtf(var + kEPS);
  h16[row * kD + t]       = (_Float16)((v0 - mu) * rs * g[t]       + bta[t]);
  h16[row * kD + t + 256] = (_Float16)((v1 - mu) * rs * g[t + 256] + bta[t + 256]);
}

// 2x4-blocked K-step: all operands in distinct registers, then 8 WMMAs.
DEVFN void gemm_step_2x4(const _Float16* A, int lda,
                         const _Float16* Bt, int ldb,
                         int k0, int lane, v8f acc[2][4]) {
  v16h a0 = load_a_frag(A + k0,            lda, lane);
  v16h a1 = load_a_frag(A + 16 * lda + k0, lda, lane);
  v16h b0 = load_bt_frag(Bt +  0 * ldb + k0, ldb, lane);
  v16h b1 = load_bt_frag(Bt + 16 * ldb + k0, ldb, lane);
  v16h b2 = load_bt_frag(Bt + 32 * ldb + k0, ldb, lane);
  v16h b3 = load_bt_frag(Bt + 48 * ldb + k0, ldb, lane);
  acc[0][0] = wmma_f16(a0, b0, acc[0][0]);
  acc[1][0] = wmma_f16(a1, b0, acc[1][0]);
  acc[0][1] = wmma_f16(a0, b1, acc[0][1]);
  acc[1][1] = wmma_f16(a1, b1, acc[1][1]);
  acc[0][2] = wmma_f16(a0, b2, acc[0][2]);
  acc[1][2] = wmma_f16(a1, b2, acc[1][2]);
  acc[0][3] = wmma_f16(a0, b3, acc[0][3]);
  acc[1][3] = wmma_f16(a1, b3, acc[1][3]);
}

// ---- Kernel 2: QKV projections. 2x4 tile block (32 rows x 64 cols)/wave ---
// q,k stored [B,H,S,DH]; v stored transposed [B,H,DH,S] (for PV GEMM B-op).
__global__ void k_qkv(const _Float16* __restrict__ h16,
                      const _Float16* __restrict__ WtAll,
                      const float* __restrict__ bq,
                      const float* __restrict__ bk,
                      const float* __restrict__ bv,
                      _Float16* __restrict__ q16,
                      _Float16* __restrict__ k16,
                      _Float16* __restrict__ vT16) {
  int lane = threadIdx.x & 31, w = threadIdx.x >> 5;
  int m0 = blockIdx.x * 32;                          // 256 m-blocks of 32 rows
  int nb = blockIdx.y * 8 + w;                       // 24 n-blocks of 64 cols
  int n0 = nb * 64;
  int mat = n0 >> 9;                                 // 0=q 1=k 2=v (64 | 512)
  int n_in0 = n0 & 511;

  const _Float16* A  = h16 + (size_t)m0 * kD;
  const _Float16* Bt = WtAll + mat * 262144 + n_in0 * 512;

  v8f acc[2][4] = {};
  for (int k0 = 0; k0 < 512; k0 += 32)
    gemm_step_2x4(A, kD, Bt, 512, k0, lane, acc);

  const float* bias = (mat == 0) ? bq : (mat == 1) ? bk : bv;
  int hi8 = (lane >> 4) << 3;
#pragma unroll
  for (int j = 0; j < 4; ++j) {
    int n_in = n_in0 + j * 16 + (lane & 15);
    float bn = bias[n_in];
    int hh = n_in >> 6, dd = n_in & 63;
#pragma unroll
    for (int i = 0; i < 2; ++i) {
#pragma unroll
      for (int r = 0; r < 8; ++r) {
        int m = m0 + i * 16 + r + hi8;
        int bidx = m >> 9, s = m & 511;
        _Float16 val = (_Float16)(acc[i][j][r] + bn);
        if (mat == 2) {
          vT16[(((bidx * kH + hh) * kDH) + dd) * kS + s] = val;
        } else {
          _Float16* dst = (mat == 0) ? q16 : k16;
          dst[(((bidx * kH + hh) * kS) + s) * kDH + dd] = val;
        }
      }
    }
  }
}

// ---- Kernel 3: fused attention per (b, h, 32-row block) -------------------
// V panel async-DMA'd to LDS overlapping: scores(WMMA) -> LDS f32 ->
// rowwise LN(lna) + mask + softmax -> probs f16 in LDS -> PV(WMMA from LDS)
// -> ao16 [B,S,D] f16.  LDS: 64K scores + 32K probs + 64K V + 128.
__global__ void k_attn(const _Float16* __restrict__ q16,
                       const _Float16* __restrict__ k16,
                       const _Float16* __restrict__ vT16,
                       const int* __restrict__ mask,
                       const float* __restrict__ lnag,
                       const float* __restrict__ lnab,
                       _Float16* __restrict__ ao16) {
  extern __shared__ char smem[];
  float*    scores    = (float*)(smem + kScoresOff);        // [32][512] f32
  _Float16* probs     = (_Float16*)(smem + kProbsOff);      // [32][512] f16
  float*    row_scale = (float*)(smem + kRowScaleOff);      // [32]
  const _Float16* vlds = (const _Float16*)(smem + kVldsOff);// [64][512] f16

  int lane = threadIdx.x & 31, w = threadIdx.x >> 5;
  int r0 = blockIdx.x * 32;            // query-row block
  int h  = blockIdx.y;
  int b  = blockIdx.z;

  const _Float16* qbh  = q16  + ((b * kH + h) * kS + r0) * kDH;
  const _Float16* kbh  = k16  + (b * kH + h) * kS * kDH;
  const _Float16* vTbh = vT16 + (b * kH + h) * kDH * kS;

  // --- Kick off async DMA of the V panel (64 KB) into LDS. ----------------
  // 256 threads x 16 chunks x 16 B; ASYNCcnt-tracked, overlaps phases A+B.
  {
    unsigned long long gbase = (unsigned long long)(uintptr_t)vTbh;
    unsigned t = threadIdx.x;
#pragma unroll
    for (int c = 0; c < 16; ++c) {
      unsigned off = (c * 256u + t) * 16u;
      unsigned lds_addr = kVldsOff + off;
      unsigned long long gaddr = gbase + off;
      asm volatile("global_load_async_to_lds_b128 %0, %1, off"
                   :: "v"(lds_addr), "v"(gaddr) : "memory");
    }
  }

  // --- Phase A: scores[32][512] = (q @ k^T) * 1/sqrt(DH) -------------------
  // 2 m-tiles x 8 n-groups (of 4 tiles); 16 groups / 8 waves = 2 each.
  // q fragment reused across 4 key tiles per K-step.
  for (int t = w; t < 16; t += 8) {
    int mt = t >> 3, ng = t & 7;
    const _Float16* A   = qbh + (mt * 16) * kDH;
    const _Float16* Bt0 = kbh + (ng * 64) * kDH;  // k rows ARE Bt for q@k^T
    v8f acc[4] = {};
#pragma unroll
    for (int k0 = 0; k0 < kDH; k0 += 32) {
      v16h a  = load_a_frag(A + k0, kDH, lane);
      v16h b0 = load_bt_frag(Bt0 +  0 * kDH + k0, kDH, lane);
      v16h b1 = load_bt_frag(Bt0 + 16 * kDH + k0, kDH, lane);
      v16h b2 = load_bt_frag(Bt0 + 32 * kDH + k0, kDH, lane);
      v16h b3 = load_bt_frag(Bt0 + 48 * kDH + k0, kDH, lane);
      acc[0] = wmma_f16(a, b0, acc[0]);
      acc[1] = wmma_f16(a, b1, acc[1]);
      acc[2] = wmma_f16(a, b2, acc[2]);
      acc[3] = wmma_f16(a, b3, acc[3]);
    }
    int hi8 = (lane >> 4) << 3;
#pragma unroll
    for (int j = 0; j < 4; ++j) {
      int ncol = ng * 64 + j * 16 + (lane & 15);
#pragma unroll
      for (int r = 0; r < 8; ++r)
        scores[(mt * 16 + r + hi8) * kS + ncol] = acc[j][r] * kQKScale;
    }
  }
  __syncthreads();

  // --- Phase B: per-row LayerNorm over keys, mask, stable softmax ----------
  // 32 rows / 8 waves = 4 rows per wave; each row is 512 f32 in LDS.
  for (int i = 0; i < 4; ++i) {
    int r = w * 4 + i;
    float* srow = scores + r * kS;
    const int* mrow = mask + ((size_t)b * kS + (r0 + r)) * kS;

    float sum = 0.f, sq = 0.f;
    for (int j = lane; j < kS; j += 32) { float v = srow[j]; sum += v; sq += v * v; }
    sum = wave_sum(sum); sq = wave_sum(sq);
    float mu  = sum * (1.0f / kS);
    float var = sq * (1.0f / kS) - mu * mu;
    float rs  = rsqrtf(var + kEPS);

    float mx = -3.4e38f;
    for (int j = lane; j < kS; j += 32) {
      float v = (srow[j] - mu) * rs * lnag[j] + lnab[j];
      if (mrow[j] == 0) v = kNEG;
      srow[j] = v;
      mx = fmaxf(mx, v);
    }
    mx = wave_max(mx);

    float es = 0.f;
    for (int j = lane; j < kS; j += 32) {
      float e = __expf(srow[j] - mx);
      es += e;
      probs[r * kS + j] = (_Float16)e;
    }
    es = wave_sum(es);
    if (lane == 0) row_scale[r] = 1.0f / es;   // fold 1/sum into PV epilogue
  }

  // V panel must be resident in LDS before phase C (each wave waits for its
  // own async loads; barrier makes every wave's chunks visible block-wide).
  asm volatile("s_wait_asynccnt 0x0" ::: "memory");
  __syncthreads();

  // --- Phase C: out[32][64] = probs @ v (both from LDS), x 1/sum -----------
  // 2 m-tiles x 4 n-tiles = 8 tiles; one per wave. K = 512.
  {
    int mt = w >> 2, nt = w & 3;
    const _Float16* A  = probs + (mt * 16) * kS;    // LDS, lda = 512
    const _Float16* Bt = vlds  + (nt * 16) * kS;    // LDS vT[d][s], ldb = 512
    v8f acc = {};
    for (int k0 = 0; k0 < kS; k0 += 32) {
      v16h a = load_a_frag(A + k0, kS, lane);
      v16h bb = load_bt_frag(Bt + k0, kS, lane);
      acc = wmma_f16(a, bb, acc);
    }
    int dcol = nt * 16 + (lane & 15);               // d within head
    int hi8 = (lane >> 4) << 3;
#pragma unroll
    for (int r = 0; r < 8; ++r) {
      int ml = mt * 16 + r + hi8;
      int s = r0 + ml;
      float val = acc[r] * row_scale[ml];
      ao16[((size_t)(b * kS + s)) * kD + h * kDH + dcol] = (_Float16)val;
    }
  }
}

// ---- Kernel 4: output projection + bias + residual, 2x4 block/wave --------
__global__ void k_proj_out(const _Float16* __restrict__ ao16,
                           const _Float16* __restrict__ WdT,   // WtAll + 3*256K
                           const float* __restrict__ bd,
                           const float* __restrict__ x,
                           float* __restrict__ out) {
  int lane = threadIdx.x & 31, w = threadIdx.x >> 5;
  int m0 = blockIdx.x * 32;            // 256 m-blocks of 32 rows
  int n0 = w * 64;                     // 8 waves cover N = 512

  const _Float16* A  = ao16 + (size_t)m0 * kD;
  const _Float16* Bt = WdT + n0 * 512;

  v8f acc[2][4] = {};
  for (int k0 = 0; k0 < 512; k0 += 32)
    gemm_step_2x4(A, kD, Bt, 512, k0, lane, acc);

  int hi8 = (lane >> 4) << 3;
#pragma unroll
  for (int j = 0; j < 4; ++j) {
    int n = n0 + j * 16 + (lane & 15);
    float bn = bd[n];
#pragma unroll
    for (int i = 0; i < 2; ++i) {
#pragma unroll
      for (int r = 0; r < 8; ++r) {
        int m = m0 + i * 16 + r + hi8;
        out[(size_t)m * kD + n] = acc[i][j][r] + bn + x[(size_t)m * kD + n];
      }
    }
  }
}

// ---------------------------------------------------------------------------
extern "C" void kernel_launch(void* const* d_in, const int* in_sizes, int n_in,
                              void* d_out, int out_size, void* d_ws, size_t ws_size,
                              hipStream_t stream) {
  const float* x    = (const float*)d_in[0];
  const int*   mask = (const int*)d_in[1];
  const float* ln_g = (const float*)d_in[2];
  const float* ln_b = (const float*)d_in[3];
  const float* lna_g = (const float*)d_in[4];
  const float* lna_b = (const float*)d_in[5];
  const float* Wq = (const float*)d_in[6];
  const float* bq = (const float*)d_in[7];
  const float* Wk = (const float*)d_in[8];
  const float* bk = (const float*)d_in[9];
  const float* Wv = (const float*)d_in[10];
  const float* bv = (const float*)d_in[11];
  const float* Wd = (const float*)d_in[12];
  const float* bd = (const float*)d_in[13];
  float* out = (float*)d_out;

  // Workspace layout (bytes); total ~42 MB.
  char* ws = (char*)d_ws;
  _Float16* WtAll = (_Float16*)(ws);                      //  4*512*512*2 = 2 MB
  _Float16* h16   = (_Float16*)(ws + (2u << 20));         //  8 MB
  _Float16* q16   = (_Float16*)(ws + (10u << 20));        //  8 MB
  _Float16* k16   = (_Float16*)(ws + (18u << 20));        //  8 MB
  _Float16* vT16  = (_Float16*)(ws + (26u << 20));        //  8 MB
  _Float16* ao16  = (_Float16*)(ws + (34u << 20));        //  8 MB

  // 0) weight transpose + f16 convert
  k_prep_weights<<<4 * 512 * 512 / 256, 256, 0, stream>>>(Wq, Wk, Wv, Wd, WtAll);
  // 1) pre-LN
  k_layernorm<<<kM, 256, 0, stream>>>(x, ln_g, ln_b, h16);
  // 2) QKV GEMM: 256 m-blocks x 24 n-blocks, 8 waves/block
  k_qkv<<<dim3(256, 3), 256, 0, stream>>>(h16, WtAll, bq, bk, bv, q16, k16, vT16);
  // 3) fused attention: (S/32, H, B) blocks, ~160 KB dynamic LDS
  k_attn<<<dim3(kS / 32, kH, kB), 256, kSmemBytes, stream>>>(
      q16, k16, vT16, mask, lna_g, lna_b, ao16);
  // 4) output projection + residual: 256 m-blocks, 8 n-blocks (1 per wave)
  k_proj_out<<<dim3(256, 1), 256, 0, stream>>>(ao16, WtAll + 3 * 262144, bd, x, out);
}
